// GAT_57389353009371
// MI455X (gfx1250) — matrix operand order repaired and checked by reading, not compiled, not run on hardware
//
#include <hip/hip_runtime.h>
#include <hip/hip_bf16.h>
#include <math.h>

// GAT for MI455X (gfx1250, wave32).
// Design: per edge type, fuse all 4 head projections into one [N,256]@[256,256]
// bf16 WMMA GEMM (v_wmma_f32_16x16x32_bf16, f32 accumulate).  Edge list is
// sorted by src with exactly 16 contiguous edges per node, so softmax and
// aggregation are per-node 16-wide ops (no atomics).  H (51MB) fits in the
// 192MB L2, so the 16-row gathers in aggregation are L2-resident.

#define GAT_ALPHA 0.2f

typedef __attribute__((ext_vector_type(16))) __bf16 v16bf;
typedef __attribute__((ext_vector_type(8)))  __bf16 v8bf;
typedef __attribute__((ext_vector_type(8)))  float  v8f;

__device__ __forceinline__ unsigned short f2bf(float f) {
  unsigned int u = __float_as_uint(f);
  unsigned int r = (u + 0x7fffu + ((u >> 16) & 1u)) >> 16;  // round-nearest-even
  return (unsigned short)r;
}
__device__ __forceinline__ float eluf(float x)  { return x > 0.f ? x : (expf(x) - 1.f); }
__device__ __forceinline__ float lreluf(float x){ return x > 0.f ? x : GAT_ALPHA * x; }

// ---- elementwise f32 -> bf16 convert --------------------------------------
__global__ void k_f32_to_bf16(const float* __restrict__ src,
                              unsigned short* __restrict__ dst, int n) {
  int t = blockIdx.x * blockDim.x + threadIdx.x;
  if (t < n) dst[t] = f2bf(src[t]);
}

// ---- weight prep: W[j] is [4,256,64] f32; produce Wt[col][k] bf16 ----------
// col = h*64 + d  (heads concatenated along columns), k = reduction dim (256)
__global__ void k_wprep(const float* __restrict__ W, unsigned short* __restrict__ Wt) {
  int t = blockIdx.x * blockDim.x + threadIdx.x;
  if (t >= 256 * 256) return;
  int c = t >> 8, kk = t & 255;
  int h = c >> 6, d = c & 63;
  Wt[(size_t)c * 256 + kk] = f2bf(W[((size_t)h * 256 + kk) * 64 + d]);
}

// ---- WMMA GEMM: H[N,256] = Xb[N,256] @ Wt^T  (Wt stored [256 cols][256 k]) -
// One wave per block; each wave computes a 16x64 output strip (4 WMMA tiles),
// reusing its A fragment 4x.  grid = (N/16, 4).
__global__ __launch_bounds__(32) void k_gemm(const unsigned short* __restrict__ Xb,
                                             const unsigned short* __restrict__ Wt,
                                             float* __restrict__ H) {
  const int lane = threadIdx.x;
  const int hi   = lane >> 4;      // 0: lanes 0-15, 1: lanes 16-31
  const int lm   = lane & 15;
  const int rowBase = blockIdx.x << 4;
  const int colBase = blockIdx.y << 6;

  const __bf16* A = (const __bf16*)Xb + (size_t)(rowBase + lm) * 256;
  const __bf16* B = (const __bf16*)Wt;

  v8f acc[4];
#pragma unroll
  for (int t = 0; t < 4; ++t)
#pragma unroll
    for (int r = 0; r < 8; ++r) acc[t][r] = 0.0f;

#pragma unroll
  for (int kb = 0; kb < 256; kb += 32) {
    // A fragment (16x32 bf16): lane<16 -> K {kb..kb+15}
    //                          lane>=16-> K {kb+16..kb+31}
    v8bf alo = *(const v8bf*)(A + kb + hi * 16);
    v8bf ahi = *(const v8bf*)(A + kb + 8 + hi * 16);
    v16bf a;
#pragma unroll
    for (int t = 0; t < 8; ++t) { a[t] = alo[t]; a[t + 8] = ahi[t]; }
#pragma unroll
    for (int t = 0; t < 4; ++t) {
      // B fragment (32x16 bf16): lanes 0-15 hold K=kb..kb+15 of col lm,
      // lanes 16-31 hold K=kb+16..kb+31 -> contiguous in column-major Wt.
      v16bf b = *(const v16bf*)(B + (size_t)(colBase + t * 16 + lm) * 256 + hi * 16 + kb);
      acc[t] = __builtin_amdgcn_wmma_f32_16x16x32_bf16(
          false, a, false, b, (short)0, acc[t], false, false);
    }
  }
  // C/D layout: VGPR r -> row r + 8*hi, col = lm
#pragma unroll
  for (int t = 0; t < 4; ++t) {
    float* out = H + (size_t)(rowBase + hi * 8) * 256 + colBase + t * 16 + lm;
#pragma unroll
    for (int r = 0; r < 8; ++r) out[(size_t)r * 256] = acc[t][r];
  }
}

// ---- attention scores: sSrc[h*N+i] = H[i, h*64:..] . a[h][:64], sDst likewise
__global__ void k_scores(const float* __restrict__ H, const float* __restrict__ a,
                         float* __restrict__ sSrc, float* __restrict__ sDst, int N) {
  int t = blockIdx.x * blockDim.x + threadIdx.x;
  if (t >= N * 4) return;
  int i = t >> 2, h = t & 3;
  const float* hr = H + (size_t)i * 256 + h * 64;
  const float* av = a + h * 128;
  float ss = 0.f, sd = 0.f;
#pragma unroll 8
  for (int d = 0; d < 64; ++d) { float v = hr[d]; ss += v * av[d]; sd += v * av[64 + d]; }
  sSrc[(size_t)h * N + i] = ss;
  sDst[(size_t)h * N + i] = sd;
}

// ---- per-node 16-wide softmax -> att[i*64 + h*16 + k] ----------------------
__global__ void k_att(const float* __restrict__ sSrc, const float* __restrict__ sDst,
                      const int* __restrict__ dstIdx, float* __restrict__ att, int N) {
  int t = blockIdx.x * blockDim.x + threadIdx.x;
  if (t >= N * 4) return;
  int i = t >> 2, h = t & 3;
  float base = sSrc[(size_t)h * N + i];
  float ev[16];
  float m = -3.0e38f;
#pragma unroll
  for (int k = 0; k < 16; ++k) {
    int d = dstIdx[(size_t)i * 16 + k];
    float e = lreluf(base + sDst[(size_t)h * N + d]);
    ev[k] = e;
    m = fmaxf(m, e);
  }
  float s = 0.f;
#pragma unroll
  for (int k = 0; k < 16; ++k) { ev[k] = expf(ev[k] - m); s += ev[k]; }
  float inv = 1.f / s;
  float* ao = att + (size_t)i * 64 + h * 16;
#pragma unroll
  for (int k = 0; k < 16; ++k) ao[k] = ev[k] * inv;
}

// ---- layer-0 aggregation: one block (256 thr) per node, col = tid ----------
// tmp[i,c] (+)= etw[j] * elu( sum_k att * H[dst_k, c] )
__global__ __launch_bounds__(256) void k_agg_l0(const float* __restrict__ H,
                                                const float* __restrict__ att,
                                                const int* __restrict__ dstIdx,
                                                float* __restrict__ tmp,
                                                const float* __restrict__ etw,
                                                int j, int add) {
  __shared__ float att_s[64];
  __shared__ int dst_s[16];
  int i = blockIdx.x, c = threadIdx.x;
  if (c < 64) att_s[c] = att[(size_t)i * 64 + c];
  if (c < 16) dst_s[c] = dstIdx[(size_t)i * 16 + c];
  __syncthreads();
  int h = c >> 6;
  float acc = 0.f;
#pragma unroll
  for (int k = 0; k < 16; ++k)
    acc += att_s[(h << 4) | k] * H[(size_t)dst_s[k] * 256 + c];
  float v = etw[j] * eluf(acc);
  size_t o = (size_t)i * 256 + c;
  if (add) tmp[o] += v; else tmp[o] = v;
}

// ---- X1 = elu(tmp + etb0), emitted as bf16 for the layer-1 GEMM ------------
__global__ void k_combine0(const float* __restrict__ tmp, const float* __restrict__ etb,
                           unsigned short* __restrict__ X1b, int n) {
  int t = blockIdx.x * blockDim.x + threadIdx.x;
  if (t < n) X1b[t] = f2bf(eluf(tmp[t] + etb[0]));
}

// ---- layer-1 aggregation: one block (64 thr) per node, mean over heads -----
__global__ __launch_bounds__(64) void k_agg_l1(const float* __restrict__ H,
                                               const float* __restrict__ att,
                                               const int* __restrict__ dstIdx,
                                               float* __restrict__ colsOut) {
  __shared__ float att_s[64];
  __shared__ int dst_s[16];
  int i = blockIdx.x, d = threadIdx.x;
  att_s[d] = att[(size_t)i * 64 + d];
  if (d < 16) dst_s[d] = dstIdx[(size_t)i * 16 + d];
  __syncthreads();
  float acc = 0.f;
#pragma unroll
  for (int h = 0; h < 4; ++h)
#pragma unroll
    for (int k = 0; k < 16; ++k)
      acc += att_s[h * 16 + k] * H[(size_t)dst_s[k] * 256 + h * 64 + d];
  colsOut[(size_t)i * 64 + d] = eluf(acc * 0.25f);
}

// ---- final edge-type mix ---------------------------------------------------
__global__ void k_final(const float* __restrict__ c0, const float* __restrict__ c1,
                        const float* __restrict__ etw, const float* __restrict__ etb,
                        float* __restrict__ out, int n) {
  int t = blockIdx.x * blockDim.x + threadIdx.x;
  if (t < n) out[t] = eluf(c0[t] * etw[0] + c1[t] * etw[1] + etb[0]);
}

extern "C" void kernel_launch(void* const* d_in, const int* in_sizes, int n_in,
                              void* d_out, int out_size, void* d_ws, size_t ws_size,
                              hipStream_t stream) {
  const float* X    = (const float*)d_in[0];
  const int*   e0   = (const int*)d_in[1];
  const int*   e1   = (const int*)d_in[2];
  const float* W0   = (const float*)d_in[3];
  const float* a0   = (const float*)d_in[4];
  const float* W1   = (const float*)d_in[5];
  const float* a1   = (const float*)d_in[6];
  const float* etw0 = (const float*)d_in[7];
  const float* etb0 = (const float*)d_in[8];
  const float* etw1 = (const float*)d_in[9];
  const float* etb1 = (const float*)d_in[10];

  const int N = in_sizes[0] / 256;   // 50000 (multiple of 16)
  const int E = in_sizes[1] / 2;     // 800000
  const int* dsts[2] = { e0 + E, e1 + E };

  size_t off = 0;
  auto carve = [&](size_t bytes) {
    void* p = (char*)d_ws + off;
    off += (bytes + 255) & ~(size_t)255;
    return p;
  };
  unsigned short* Xb  = (unsigned short*)carve((size_t)N * 256 * 2);  // X / X1 bf16
  unsigned short* Wt  = (unsigned short*)carve((size_t)256 * 256 * 2);
  float* H    = (float*)carve((size_t)N * 256 * 4);
  float* Tmp  = (float*)carve((size_t)N * 256 * 4);  // layer-0 accum / layer-1 cols
  float* sSrc = (float*)carve((size_t)N * 4 * 4);
  float* sDst = (float*)carve((size_t)N * 4 * 4);
  float* Att  = (float*)carve((size_t)N * 64 * 4);

  const int nXc = N * 256;
  dim3 b256(256), b64(64), b32(32);

  k_f32_to_bf16<<<(nXc + 255) / 256, b256, 0, stream>>>(X, Xb, nXc);

  // ---- layer 0 ----
  for (int j = 0; j < 2; ++j) {
    k_wprep<<<(65536 + 255) / 256, b256, 0, stream>>>(W0 + (size_t)j * 4 * 256 * 64, Wt);
    k_gemm<<<dim3(N / 16, 4), b32, 0, stream>>>(Xb, Wt, H);
    k_scores<<<(N * 4 + 255) / 256, b256, 0, stream>>>(H, a0 + (size_t)j * 512, sSrc, sDst, N);
    k_att<<<(N * 4 + 255) / 256, b256, 0, stream>>>(sSrc, sDst, dsts[j], Att, N);
    k_agg_l0<<<N, b256, 0, stream>>>(H, Att, dsts[j], Tmp, etw0, j, j);
  }
  k_combine0<<<(nXc + 255) / 256, b256, 0, stream>>>(Tmp, etb0, Xb, nXc);

  // ---- layer 1 (cols reuse Tmp region: 2 * N*64 f32 <= N*256 f32) ----
  float* cols = Tmp;
  for (int j = 0; j < 2; ++j) {
    k_wprep<<<(65536 + 255) / 256, b256, 0, stream>>>(W1 + (size_t)j * 4 * 256 * 64, Wt);
    k_gemm<<<dim3(N / 16, 4), b32, 0, stream>>>(Xb, Wt, H);
    k_scores<<<(N * 4 + 255) / 256, b256, 0, stream>>>(H, a1 + (size_t)j * 512, sSrc, sDst, N);
    k_att<<<(N * 4 + 255) / 256, b256, 0, stream>>>(sSrc, sDst, dsts[j], Att, N);
    k_agg_l1<<<N, b64, 0, stream>>>(H, Att, dsts[j], cols + (size_t)j * N * 64);
  }
  k_final<<<(N * 64 + 255) / 256, b256, 0, stream>>>(cols, cols + (size_t)N * 64,
                                                     etw1, etb1, (float*)d_out, N * 64);
}
